// TGCNWrapper_64364379898158
// MI455X (gfx1250) — matrix operand order repaired and checked by reading, not compiled, not run on hardware
//
#include <hip/hip_runtime.h>

// ---------------------------------------------------------------------------
// TGCN (2-layer graph GRU), B=64, T=500, N=21, HID=128.
// Design: 1 workgroup per batch element; recurrence state in registers/LDS;
// all matmuls via v_wmma_f32_16x16x32_bf16 (wave32); weights pre-packed into
// WMMA B-fragment layout with contiguous k (16*hi + i) in workspace (L2-resident, broadcast to WGs).
// ---------------------------------------------------------------------------

typedef unsigned short u16;
typedef __attribute__((ext_vector_type(16))) __bf16 v16bf;
typedef __attribute__((ext_vector_type(8)))  float  v8f;
typedef __attribute__((ext_vector_type(16))) u16    v16u;
typedef __attribute__((ext_vector_type(8)))  u16    v8u;

#define B_   64
#define T_   500
#define N_   21
#define HID_ 128
#define E_   210

// f32 -> bf16 bits, round-to-nearest-even
static __device__ __forceinline__ u16 f2b(float f) {
    unsigned u = __float_as_uint(f);
    unsigned r = u + 0x7FFFu + ((u >> 16) & 1u);
    return (u16)(r >> 16);
}

static __device__ __forceinline__ float sigm(float x) {
    return 1.0f / (1.0f + __expf(-x));
}
static __device__ __forceinline__ float tanh_fast(float x) {
    float e = __expf(2.0f * x);          // safe at +/-inf
    return 1.0f - 2.0f / (e + 1.0f);
}

static __device__ __forceinline__ v8f vzero8() {
    v8f z;
#pragma unroll
    for (int i = 0; i < 8; ++i) z[i] = 0.0f;
    return z;
}

// D = A*B + C, bf16 16x16x32, f32 accum
static __device__ __forceinline__ v8f wmma_bf16(v16bf a, v16bf b, v8f c) {
    return __builtin_amdgcn_wmma_f32_16x16x32_bf16(false, a, false, b,
                                                   (short)0, c, false, false);
}

// A-fragment (16x32) from row-major bf16 buffer (stride halves).
// lane: m=(lane&15)+mtbase, hi=lane>>4; h[0..7]=A[m][kbase+16hi..], h[8..15]=A[m][kbase+16hi+8..]
static __device__ __forceinline__ v16bf loadA(const u16* buf, int stride,
                                              int mtbase, int kbase, int lane) {
    int m  = (lane & 15) + mtbase;
    int hi = lane >> 4;
    const u16* p0 = buf + m * stride + kbase + 16 * hi;
    union { v16bf b; v8u h[2]; } r;
    r.h[0] = *(const v8u*)p0;
    r.h[1] = *(const v8u*)(p0 + 8);
    return r.b;
}

// B-fragment (32x16) from pre-packed global fragments: one 32B load per lane.
static __device__ __forceinline__ v16bf loadBpk(const u16* pk, int kc, int nt, int lane) {
    union { v16bf b; v16u u; } r;
    r.u = *(const v16u*)(pk + ((((kc * 8) + nt) * 32 + lane) << 4));
    return r.b;
}

// B-fragment from transposed LDS buffer Yt[col][32] (h[i] = Y[16hi+i][col]).
static __device__ __forceinline__ v16bf loadBt(const u16* yt, int nt, int lane) {
    int n = lane & 15, hi = lane >> 4;
    union { v16bf b; v16u u; } r;
    r.u = *(const v16u*)(yt + (nt * 16 + n) * 32 + 16 * hi);
    return r.b;
}

// ---------------------------------------------------------------------------
// Prep kernel 1: PyG gcn_norm -> dense normalized adjacency, f32 [32][32] (padded w/ 0)
// ---------------------------------------------------------------------------
__global__ void build_ahat_kernel(const int* __restrict__ ei,
                                  const float* __restrict__ ew,
                                  float* __restrict__ ahat) {
    __shared__ float deg[32];
    __shared__ float dinv[32];
    __shared__ float dense[32 * 32];
    int t = threadIdx.x;  // 256 threads, 1 block
    for (int i = t; i < 1024; i += 256) dense[i] = 0.0f;
    if (t < 32) deg[t] = 0.0f;
    __syncthreads();
    if (t < E_) atomicAdd(&deg[ei[E_ + t]], ew[t]);   // dst-degree
    if (t < N_) atomicAdd(&deg[t], 1.0f);             // self loops, weight 1
    __syncthreads();
    if (t < 32) dinv[t] = (deg[t] > 0.0f) ? rsqrtf(deg[t]) : 0.0f;
    __syncthreads();
    if (t < E_) {
        int s = ei[t], d = ei[E_ + t];
        atomicAdd(&dense[d * 32 + s], dinv[s] * ew[t] * dinv[d]);
    }
    if (t < N_) atomicAdd(&dense[t * 32 + t], dinv[t] * dinv[t]);
    __syncthreads();
    for (int i = t; i < 1024; i += 256) ahat[i] = dense[i];
}

// ---------------------------------------------------------------------------
// Prep kernel 2: pack a K x 128 f32 weight into bf16 WMMA B-fragments:
// dst[((kc*8+nt)*32+lane)*16 + i] = bf16( W[kc*32 + 16*(lane>>4) + i][nt*16 + (lane&15)] )
// ---------------------------------------------------------------------------
__global__ void pack_wmma_b_kernel(const float* __restrict__ W,
                                   u16* __restrict__ dst, int K) {
    int total = K * 128;
    for (int idx = blockIdx.x * blockDim.x + threadIdx.x; idx < total;
         idx += gridDim.x * blockDim.x) {
        int i    = idx & 15;
        int lane = (idx >> 4) & 31;
        int f    = idx >> 9;
        int kc = f >> 3, nt = f & 7;
        int n = lane & 15, hi = lane >> 4;
        int kk  = kc * 32 + 16 * hi + i;
        int col = nt * 16 + n;
        dst[idx] = f2b(W[kk * 128 + col]);
    }
}

// ---------------------------------------------------------------------------
// Main recurrent kernel: 64 blocks (one per batch element) x 256 threads (8 waves)
// ---------------------------------------------------------------------------
struct Params {
    const float* x;
    const float *Wz1, *bz1, *lzb1;
    const float *Wr1, *br1, *lrb1;
    const float *Wh1, *bh1, *lhb1;
    const float *bz2, *lzb2, *br2, *lrb2, *bh2, *lhb2;
    const float *clsw, *clsb;
    const float* ahat;
    const u16 *pkz1, *pkr1, *pkh1;   // lzw_1/lrw_1/lhw_1 packed (K=256)
    const u16 *pkz2, *pkr2, *pkh2;   // lzw_2/lrw_2/lhw_2 packed (K=256)
    const u16 *pwz2, *pwr2, *pwh2;   // Wz_2/Wr_2/Wh_2 packed (K=128)
    float* out;
};

__global__ __launch_bounds__(256) void tgcn_main_kernel(Params p) {
    __shared__ float ahatL[32 * 32];
    __shared__ float xsh[32];
    __shared__ float axsh[32];
    __shared__ u16 cat1[32 * 256];       // [G1 | H1] bf16, row-major (A operand)
    __shared__ u16 cat2[32 * 256];       // [G2 | H2] bf16, row-major (A operand)
    __shared__ u16 h1rm[32 * 128];       // H1 bf16 row-major (A operand, cell2 Y)
    __shared__ u16 yt[3][128 * 32];      // Y transposed (B operand for adjacency)
    __shared__ float red[256];

    const int tid  = threadIdx.x;
    const int lane = tid & 31;
    const int w    = tid >> 5;           // wave id = n-tile (0..7)
    const int b    = blockIdx.x;
    const int n    = lane & 15;
    const int hi   = lane >> 4;
    const int ncol = w * 16 + n;         // HID column owned by this lane

    // init LDS (H halves must start as zeros == H0)
    for (int i = tid; i < 32 * 256; i += 256) { cat1[i] = 0; cat2[i] = 0; }
    for (int i = tid; i < 32 * 128; i += 256) h1rm[i] = 0;
    for (int i = tid; i < 1024; i += 256) ahatL[i] = p.ahat[i];
    __syncthreads();

    // A_hat A-fragments (K=32, padded cols are zero)
    v16bf ahatFrag[2];
#pragma unroll
    for (int mt = 0; mt < 2; ++mt) {
        int m = (lane & 15) + 16 * mt;
        union { v16bf bf; u16 u[16]; } r;
#pragma unroll
        for (int i = 0; i < 8; ++i) {
            r.u[i]     = f2b(ahatL[m * 32 + 16 * hi + i]);
            r.u[8 + i] = f2b(ahatL[m * 32 + 16 * hi + 8 + i]);
        }
        ahatFrag[mt] = r.bf;
    }

    // per-lane scalar params for owned column
    const float W1g[3] = { p.Wz1[ncol], p.Wr1[ncol], p.Wh1[ncol] };
    const float b1g[3] = { p.bz1[ncol], p.br1[ncol], p.bh1[ncol] };
    const float l1b[3] = { p.lzb1[ncol], p.lrb1[ncol], p.lhb1[ncol] };
    const float b2g[3] = { p.bz2[ncol], p.br2[ncol], p.bh2[ncol] };
    const float l2b[3] = { p.lzb2[ncol], p.lrb2[ncol], p.lhb2[ncol] };
    const u16* pk1[3] = { p.pkz1, p.pkr1, p.pkh1 };
    const u16* pk2[3] = { p.pkz2, p.pkr2, p.pkh2 };
    const u16* pw2[3] = { p.pwz2, p.pwr2, p.pwh2 };
    const float clsw = p.clsw[ncol];

    v8f H1f[2], H2f[2];
    H1f[0] = vzero8(); H1f[1] = vzero8();
    H2f[0] = vzero8(); H2f[1] = vzero8();
    float accOut = 0.0f;

    for (int t = 0; t < T_; ++t) {
        // ---- x_t, ax = A_hat @ x_t (rank-1 trick for cell 1 GCN) ----
        if (tid < 32)
            xsh[tid] = (tid < N_) ? p.x[(size_t)b * T_ * N_ + (size_t)t * N_ + tid] : 0.0f;
        __syncthreads();
        if (tid < 32) {
            float ax = 0.0f;
#pragma unroll
            for (int k = 0; k < N_; ++k) ax += ahatL[tid * 32 + k] * xsh[k];
            axsh[tid] = ax;
        }
        __syncthreads();

        // ================= Cell 1 (X = x_t, H = H1) =================
        v8f Z1[2], R1[2];
#pragma unroll
        for (int g = 0; g < 2; ++g) {   // z, r gates
#pragma unroll
            for (int r2 = 0; r2 < 16; ++r2) {   // G1 = ax*W1 + b1 into cat1[:,0:128]
                int row = hi * 16 + r2;
                float gv = (row < N_) ? axsh[row] * W1g[g] + b1g[g] : 0.0f;
                cat1[row * 256 + ncol] = f2b(gv);
            }
            __syncthreads();
#pragma unroll
            for (int mt = 0; mt < 2; ++mt) {
                v8f acc = vzero8();
#pragma unroll
                for (int kc = 0; kc < 8; ++kc)
                    acc = wmma_bf16(loadA(cat1, 256, mt * 16, kc * 32, lane),
                                    loadBpk(pk1[g], kc, w, lane), acc);
                v8f r;
#pragma unroll
                for (int v = 0; v < 8; ++v) r[v] = sigm(acc[v] + l1b[g]);
                if (g == 0) Z1[mt] = r; else R1[mt] = r;
            }
            __syncthreads();
        }
        // h gate: H-half := H1*R1 (masked), G-half := Gh
#pragma unroll
        for (int mt = 0; mt < 2; ++mt)
#pragma unroll
            for (int v = 0; v < 8; ++v) {
                int row = mt * 16 + 8 * hi + v;
                cat1[row * 256 + 128 + ncol] =
                    (row < N_) ? f2b(H1f[mt][v] * R1[mt][v]) : (u16)0;
            }
#pragma unroll
        for (int r2 = 0; r2 < 16; ++r2) {
            int row = hi * 16 + r2;
            float gv = (row < N_) ? axsh[row] * W1g[2] + b1g[2] : 0.0f;
            cat1[row * 256 + ncol] = f2b(gv);
        }
        __syncthreads();
#pragma unroll
        for (int mt = 0; mt < 2; ++mt) {
            v8f acc = vzero8();
#pragma unroll
            for (int kc = 0; kc < 8; ++kc)
                acc = wmma_bf16(loadA(cat1, 256, mt * 16, kc * 32, lane),
                                loadBpk(pk1[2], kc, w, lane), acc);
#pragma unroll
            for (int v = 0; v < 8; ++v) {
                float ht = tanh_fast(acc[v] + l1b[2]);
                float z  = Z1[mt][v];
                H1f[mt][v] = z * H1f[mt][v] + (1.0f - z) * ht;
            }
        }
        __syncthreads();   // all reads of cat1 done before H-half overwrite
#pragma unroll
        for (int mt = 0; mt < 2; ++mt)
#pragma unroll
            for (int v = 0; v < 8; ++v) {
                int row = mt * 16 + 8 * hi + v;
                u16 hb = (row < N_) ? f2b(H1f[mt][v]) : (u16)0;
                h1rm[row * 128 + ncol] = hb;
                cat1[row * 256 + 128 + ncol] = hb;   // H1 for next step
            }
        __syncthreads();

        // ================= Cell 2 (X = H1, H = H2) =================
        // Yg = H1 @ W2g  (store transposed for adjacency B operand)
#pragma unroll
        for (int g = 0; g < 3; ++g)
#pragma unroll
            for (int mt = 0; mt < 2; ++mt) {
                v8f acc = vzero8();
#pragma unroll
                for (int kc = 0; kc < 4; ++kc)
                    acc = wmma_bf16(loadA(h1rm, 128, mt * 16, kc * 32, lane),
                                    loadBpk(pw2[g], kc, w, lane), acc);
                v8u val;
#pragma unroll
                for (int v = 0; v < 8; ++v) val[v] = f2b(acc[v]);
                *(v8u*)&yt[g][ncol * 32 + mt * 16 + 8 * hi] = val;
            }
        __syncthreads();

        v8f Z2[2], R2[2];
#pragma unroll
        for (int g = 0; g < 2; ++g) {   // z, r gates
#pragma unroll
            for (int mt = 0; mt < 2; ++mt) {       // G2 = A_hat @ Yg + b2
                v8f acc = wmma_bf16(ahatFrag[mt], loadBt(&yt[g][0], w, lane), vzero8());
#pragma unroll
                for (int v = 0; v < 8; ++v) {
                    int row = mt * 16 + 8 * hi + v;
                    cat2[row * 256 + ncol] = f2b(acc[v] + b2g[g]);
                }
            }
            __syncthreads();
#pragma unroll
            for (int mt = 0; mt < 2; ++mt) {
                v8f acc = vzero8();
#pragma unroll
                for (int kc = 0; kc < 8; ++kc)
                    acc = wmma_bf16(loadA(cat2, 256, mt * 16, kc * 32, lane),
                                    loadBpk(pk2[g], kc, w, lane), acc);
                v8f r;
#pragma unroll
                for (int v = 0; v < 8; ++v) r[v] = sigm(acc[v] + l2b[g]);
                if (g == 0) Z2[mt] = r; else R2[mt] = r;
            }
            __syncthreads();
        }
        // h gate: H-half := H2*R2 (masked), G-half := Gh2
#pragma unroll
        for (int mt = 0; mt < 2; ++mt)
#pragma unroll
            for (int v = 0; v < 8; ++v) {
                int row = mt * 16 + 8 * hi + v;
                cat2[row * 256 + 128 + ncol] =
                    (row < N_) ? f2b(H2f[mt][v] * R2[mt][v]) : (u16)0;
            }
#pragma unroll
        for (int mt = 0; mt < 2; ++mt) {
            v8f acc = wmma_bf16(ahatFrag[mt], loadBt(&yt[2][0], w, lane), vzero8());
#pragma unroll
            for (int v = 0; v < 8; ++v) {
                int row = mt * 16 + 8 * hi + v;
                cat2[row * 256 + ncol] = f2b(acc[v] + b2g[2]);
            }
        }
        __syncthreads();
#pragma unroll
        for (int mt = 0; mt < 2; ++mt) {
            v8f acc = vzero8();
#pragma unroll
            for (int kc = 0; kc < 8; ++kc)
                acc = wmma_bf16(loadA(cat2, 256, mt * 16, kc * 32, lane),
                                loadBpk(pk2[2], kc, w, lane), acc);
#pragma unroll
            for (int v = 0; v < 8; ++v) {
                float ht = tanh_fast(acc[v] + l2b[2]);
                float z  = Z2[mt][v];
                float hn = z * H2f[mt][v] + (1.0f - z) * ht;
                H2f[mt][v] = hn;
                int row = mt * 16 + 8 * hi + v;
                if (row < N_) accOut += hn;    // sum over nodes & time
            }
        }
        __syncthreads();   // all reads of cat2 done before H-half overwrite
#pragma unroll
        for (int mt = 0; mt < 2; ++mt)
#pragma unroll
            for (int v = 0; v < 8; ++v) {
                int row = mt * 16 + 8 * hi + v;
                cat2[row * 256 + 128 + ncol] =
                    (row < N_) ? f2b(H2f[mt][v]) : (u16)0;
            }
        __syncthreads();
    }

    // out[b] = (sum_{t,m,n} H2 * clsw[n]) / (T*N) + clsb
    red[tid] = accOut * clsw;
    __syncthreads();
    if (tid == 0) {
        float s = 0.0f;
        for (int i = 0; i < 256; ++i) s += red[i];
        p.out[b] = s / (float)(T_ * N_) + p.clsb[0];
    }
}

// ---------------------------------------------------------------------------
// Host side
// ---------------------------------------------------------------------------
extern "C" void kernel_launch(void* const* d_in, const int* in_sizes, int n_in,
                              void* d_out, int out_size, void* d_ws, size_t ws_size,
                              hipStream_t stream) {
    (void)in_sizes; (void)n_in; (void)out_size; (void)ws_size;

    const float* x  = (const float*)d_in[0];
    const int*   ei = (const int*)d_in[1];
    const float* ew = (const float*)d_in[2];
    const float* Wz1 = (const float*)d_in[3];  const float* bz1 = (const float*)d_in[4];
    const float* lzw1 = (const float*)d_in[5]; const float* lzb1 = (const float*)d_in[6];
    const float* Wr1 = (const float*)d_in[7];  const float* br1 = (const float*)d_in[8];
    const float* lrw1 = (const float*)d_in[9]; const float* lrb1 = (const float*)d_in[10];
    const float* Wh1 = (const float*)d_in[11]; const float* bh1 = (const float*)d_in[12];
    const float* lhw1 = (const float*)d_in[13];const float* lhb1 = (const float*)d_in[14];
    const float* Wz2 = (const float*)d_in[15]; const float* bz2 = (const float*)d_in[16];
    const float* lzw2 = (const float*)d_in[17];const float* lzb2 = (const float*)d_in[18];
    const float* Wr2 = (const float*)d_in[19]; const float* br2 = (const float*)d_in[20];
    const float* lrw2 = (const float*)d_in[21];const float* lrb2 = (const float*)d_in[22];
    const float* Wh2 = (const float*)d_in[23]; const float* bh2 = (const float*)d_in[24];
    const float* lhw2 = (const float*)d_in[25];const float* lhb2 = (const float*)d_in[26];
    const float* clsw = (const float*)d_in[27];const float* clsb = (const float*)d_in[28];

    // workspace layout: [ ahat f32 32x32 : 4096B ][ packed bf16 fragments ]
    char* ws = (char*)d_ws;
    float* ahat = (float*)ws;
    u16* pkbase = (u16*)(ws + 4096);
    u16* pkz1 = pkbase;             // 256x128 -> 32768 halfs each
    u16* pkr1 = pkbase + 32768;
    u16* pkh1 = pkbase + 65536;
    u16* pkz2 = pkbase + 98304;
    u16* pkr2 = pkbase + 131072;
    u16* pkh2 = pkbase + 163840;
    u16* pwz2 = pkbase + 196608;    // 128x128 -> 16384 halfs each
    u16* pwr2 = pkbase + 212992;
    u16* pwh2 = pkbase + 229376;    // total ~496KB of workspace used

    build_ahat_kernel<<<1, 256, 0, stream>>>(ei, ew, ahat);
    pack_wmma_b_kernel<<<64, 256, 0, stream>>>(lzw1, pkz1, 256);
    pack_wmma_b_kernel<<<64, 256, 0, stream>>>(lrw1, pkr1, 256);
    pack_wmma_b_kernel<<<64, 256, 0, stream>>>(lhw1, pkh1, 256);
    pack_wmma_b_kernel<<<64, 256, 0, stream>>>(lzw2, pkz2, 256);
    pack_wmma_b_kernel<<<64, 256, 0, stream>>>(lrw2, pkr2, 256);
    pack_wmma_b_kernel<<<64, 256, 0, stream>>>(lhw2, pkh2, 256);
    pack_wmma_b_kernel<<<64, 256, 0, stream>>>(Wz2, pwz2, 128);
    pack_wmma_b_kernel<<<64, 256, 0, stream>>>(Wr2, pwr2, 128);
    pack_wmma_b_kernel<<<64, 256, 0, stream>>>(Wh2, pwh2, 128);

    Params p;
    p.x = x;
    p.Wz1 = Wz1; p.bz1 = bz1; p.lzb1 = lzb1;
    p.Wr1 = Wr1; p.br1 = br1; p.lrb1 = lrb1;
    p.Wh1 = Wh1; p.bh1 = bh1; p.lhb1 = lhb1;
    p.bz2 = bz2; p.lzb2 = lzb2; p.br2 = br2; p.lrb2 = lrb2; p.bh2 = bh2; p.lhb2 = lhb2;
    p.clsw = clsw; p.clsb = clsb;
    p.ahat = ahat;
    p.pkz1 = pkz1; p.pkr1 = pkr1; p.pkh1 = pkh1;
    p.pkz2 = pkz2; p.pkr2 = pkr2; p.pkh2 = pkh2;
    p.pwz2 = pwz2; p.pwr2 = pwr2; p.pwh2 = pwh2;
    p.out = (float*)d_out;

    tgcn_main_kernel<<<B_, 256, 0, stream>>>(p);
}
